// ConsolidationModel_77068893160212
// MI455X (gfx1250) — compile-verified, matrix operand order checked by reading, not run on hardware
//
#include <hip/hip_runtime.h>

// ============================================================================
// ConsolidationModel — dead-code-eliminated implementation for gfx1250.
//
// The reference's scan schedule is data-independent; consolidations fire at
// t=15,31,47 and each result is fully shifted out of the M=8 ring before the
// next use. Final buffer == embed[seqs[:,55:63]], count==8. Therefore:
//   h   = [embed[query_tok], mean_{j=55..62} embed[seqs[:,j]]]   (B,128)
//   out = relu(h @ r1_w.T + r1_b) @ r2_w.T + r2_b                (B,64)
// Both GEMMs run on v_wmma_f32_16x16x32_f16 (16 batch rows per wave tile,
// f32 accumulate), with a per-wave LDS bounce to convert GEMM1's C/D
// fragment layout into GEMM2's A fragment layout.
// ============================================================================

typedef __attribute__((ext_vector_type(16))) _Float16 v16h;
typedef __attribute__((ext_vector_type(8)))  float    v8f;

#define HDIM  64
#define SEQL  64
#define BATCH 2048
#define TILES (BATCH / 16)   // 128 wave-tiles of 16 rows

// 16-bit A-matrix 16x32 packing (ISA 7.12.2): lane L holds row M=L%16,
// khalf=L/16; element e of the v16h maps to k = 16*(e>>3) + 8*khalf + (e&7).
// We assume B (32x16, lane = column N) uses the symmetric packing.
__device__ __forceinline__ int frag_k(int e, int khalf) {
    return 16 * (e >> 3) + 8 * khalf + (e & 7);
}

// ----------------------------------------------------------------------------
// Prep: convert r1_w (64x128 f32) and r2_w (64x64 f32) into pre-swizzled f16
// WMMA B-fragments in workspace.
//   wsB1: [nt(4)][kk(4)][lane(32)][e(16)]  value = r1_w[nt*16+lane%16][kk*32+frag_k]
//   wsB2: [nt(4)][kk(2)][lane(32)][e(16)]  value = r2_w[nt*16+lane%16][kk*32+frag_k]
// ----------------------------------------------------------------------------
__global__ void prep_weights(const float* __restrict__ r1w,
                             const float* __restrict__ r2w,
                             _Float16* __restrict__ wsB1,
                             _Float16* __restrict__ wsB2) {
    int idx = blockIdx.x * blockDim.x + threadIdx.x;
    if (idx < 4 * 4 * 32 * 16) {                       // 8192 B1 elements
        int e = idx & 15, lane = (idx >> 4) & 31, kk = (idx >> 9) & 3, nt = idx >> 11;
        int n = nt * 16 + (lane & 15);
        int k = kk * 32 + frag_k(e, lane >> 4);
        wsB1[idx] = (_Float16)r1w[n * 128 + k];
    } else if (idx < 8192 + 4 * 2 * 32 * 16) {         // 4096 B2 elements
        int j = idx - 8192;
        int e = j & 15, lane = (j >> 4) & 31, kk = (j >> 9) & 1, nt = j >> 10;
        int n = nt * 16 + (lane & 15);
        int k = kk * 32 + frag_k(e, lane >> 4);
        wsB2[j] = (_Float16)r2w[n * 64 + k];
    }
}

// ----------------------------------------------------------------------------
// Main: one wave (32 lanes) per 16-row batch tile; 8 waves / block.
// ----------------------------------------------------------------------------
__global__ __launch_bounds__(256) void consolidation_head(
    const float* __restrict__ embed,   // (66, 64)
    const float* __restrict__ r1b,     // (64,)
    const float* __restrict__ r2b,     // (64,)
    const int*   __restrict__ seqs,    // (B, 64)
    const int*   __restrict__ qtok,    // (B,)
    const _Float16* __restrict__ wsB1,
    const _Float16* __restrict__ wsB2,
    float* __restrict__ out)           // (B, 64)
{
    __shared__ _Float16 lds_hid[8][16 * HDIM];   // per-wave 16x64 f16 bounce

    const int lane  = threadIdx.x & 31;
    const int wave  = threadIdx.x >> 5;
    const int tile  = blockIdx.x * 8 + wave;
    const int m     = lane & 15;     // A-fragment row within tile
    const int khalf = lane >> 4;
    const int r     = tile * 16 + m; // batch row owned by this lane's A rows

    // Warm the (tiny, heavily reused) weight fragments into cache.
    __builtin_prefetch(wsB1 + (size_t)lane * 256, 0, 0);
    __builtin_prefetch(wsB2 + (size_t)lane * 128, 0, 0);

    // Gather bases: query-token row and the 8 surviving buffer tokens.
    const int qrow = qtok[r] * HDIM;
    int trow[8];
#pragma unroll
    for (int j = 0; j < 8; ++j) trow[j] = seqs[r * SEQL + 55 + j] * HDIM;

    // Build GEMM1 A fragments: h (16x128) in 4 K-chunks of 32.
    // k < 64  -> embed[query_tok], k >= 64 -> mean of 8 token embeddings.
    v16h a1[4];
#pragma unroll
    for (int kk = 0; kk < 4; ++kk) {
#pragma unroll
        for (int e = 0; e < 16; ++e) {
            int k = kk * 32 + frag_k(e, khalf);
            float val;
            if (k < HDIM) {
                val = embed[qrow + k];
            } else {
                int k2 = k - HDIM;
                float s = 0.f;
#pragma unroll
                for (int j = 0; j < 8; ++j) s += embed[trow[j] + k2];
                val = s * 0.125f;
            }
            a1[kk][e] = (_Float16)val;
        }
    }

    // GEMM1: hid = relu(h @ r1_w.T + r1_b), 16x64, fused bias+ReLU on C/D frag.
    // C/D layout: VGPR v, lane L -> row v + 8*(L/16), col L%16.
#pragma unroll
    for (int nt = 0; nt < 4; ++nt) {
        v8f c = {};
#pragma unroll
        for (int kk = 0; kk < 4; ++kk) {
            v16h b = *(const v16h*)(wsB1 + ((nt * 4 + kk) * 32 + lane) * 16);
            c = __builtin_amdgcn_wmma_f32_16x16x32_f16(
                    false, a1[kk], false, b, (short)0, c, false, false);
        }
        int   n    = nt * 16 + (lane & 15);
        float bias = r1b[n];
#pragma unroll
        for (int v = 0; v < 8; ++v) {
            float x = c[v] + bias;
            x = x > 0.f ? x : 0.f;
            lds_hid[wave][(v + 8 * khalf) * HDIM + n] = (_Float16)x;
        }
    }

    __syncthreads();   // make the C/D-layout stores visible for A-layout reads

    // Re-read hid from LDS in A-fragment layout: 16x64 in 2 K-chunks.
    v16h a2[2];
#pragma unroll
    for (int kk = 0; kk < 2; ++kk)
#pragma unroll
        for (int e = 0; e < 16; ++e)
            a2[kk][e] = lds_hid[wave][m * HDIM + kk * 32 + frag_k(e, khalf)];

    // GEMM2: out = hid @ r2_w.T + r2_b, 16x64, store f32.
#pragma unroll
    for (int nt = 0; nt < 4; ++nt) {
        v8f c = {};
#pragma unroll
        for (int kk = 0; kk < 2; ++kk) {
            v16h b = *(const v16h*)(wsB2 + ((nt * 2 + kk) * 32 + lane) * 16);
            c = __builtin_amdgcn_wmma_f32_16x16x32_f16(
                    false, a2[kk], false, b, (short)0, c, false, false);
        }
        int   n    = nt * 16 + (lane & 15);
        float bias = r2b[n];
#pragma unroll
        for (int v = 0; v < 8; ++v)
            out[(size_t)(tile * 16 + v + 8 * khalf) * HDIM + n] = c[v] + bias;
    }
}

extern "C" void kernel_launch(void* const* d_in, const int* in_sizes, int n_in,
                              void* d_out, int out_size, void* d_ws, size_t ws_size,
                              hipStream_t stream) {
    // setup_inputs() order: 0 embed, ..., 18 r1_w, 19 r1_b, 20 r2_w, 21 r2_b,
    // 22 seqs (int32), 23 query_tok (int32).
    const float* embed = (const float*)d_in[0];
    const float* r1w   = (const float*)d_in[18];
    const float* r1b   = (const float*)d_in[19];
    const float* r2w   = (const float*)d_in[20];
    const float* r2b   = (const float*)d_in[21];
    const int*   seqs  = (const int*)d_in[22];
    const int*   qtok  = (const int*)d_in[23];
    float*       out   = (float*)d_out;

    _Float16* wsB1 = (_Float16*)d_ws;       // 8192 halfs (16 KB)
    _Float16* wsB2 = wsB1 + 8192;           // 4096 halfs ( 8 KB)

    prep_weights<<<48, 256, 0, stream>>>(r1w, r2w, wsB1, wsB2);            // 12288 threads
    consolidation_head<<<TILES / 8, 256, 0, stream>>>(embed, r1b, r2b,
                                                      seqs, qtok, wsB1, wsB2, out);
}
